// MultiHeadAttention_18047452578106
// MI455X (gfx1250) — matrix operand order repaired and checked by reading, not compiled, not run on hardware
//
#include <hip/hip_runtime.h>
#include <hip/hip_bf16.h>
#include <math.h>
#include <stdint.h>

// Problem constants (from reference)
#define BQ   2
#define SEQ  2048
#define DM   1024
#define NH   16
#define DH   64

typedef __bf16 bf16_t;
typedef __attribute__((ext_vector_type(16))) __bf16 v16bf;
typedef __attribute__((ext_vector_type(8)))  float  v8f;

// ---------------------------------------------------------------------------
// WMMA wrapper: D(f32 16x16) = A(bf16 16x32) x B(bf16 32x16) + C
// ---------------------------------------------------------------------------
__device__ __forceinline__ v8f wmma_bf16(v16bf a, v16bf b, v8f c) {
  return __builtin_amdgcn_wmma_f32_16x16x32_bf16(
      /*neg_a=*/false, a, /*neg_b=*/false, b,
      /*c_mod=*/(short)0, c, /*reuse_a=*/false, /*reuse_b=*/false);
}

// A fragment (16 rows x 32 K), row-major in LDS with stride lda (elements).
// ISA 7.12.2 (16-bit A 16x32): lane L (L<16) row M=L holds K = {kg*8..kg*8+7,
// 16+kg*8..16+kg*8+7} with kg = lane>>4.  Two 16B contiguous runs per lane.
__device__ __forceinline__ v16bf ld_frag_a(const bf16_t* base, int lda) {
  const int lane = threadIdx.x & 31;
  const int row  = lane & 15;
  const int kg   = lane >> 4;
  const bf16_t* p0 = base + row * lda + kg * 8;
  const bf16_t* p1 = p0 + 16;
  v16bf a;
#pragma unroll
  for (int j = 0; j < 8; ++j) { a[j] = p0[j]; a[8 + j] = p1[j]; }
  return a;
}

// B fragment (32 K x 16 cols). Stored K-CONTIGUOUS in LDS: base points at
// (col0,k0) of a [16 col][32 k] region with stride ldt. Lane holds col
// n = lane&15, K = {kg*8..kg*8+7, 16+kg*8..16+kg*8+7} with kg = lane>>4.
__device__ __forceinline__ v16bf ld_frag_bt(const bf16_t* base, int ldt) {
  const int lane = threadIdx.x & 31;
  const int n  = lane & 15;
  const int kg = lane >> 4;
  const bf16_t* p = base + n * ldt + kg * 8;
  v16bf b;
#pragma unroll
  for (int j = 0; j < 8; ++j) { b[j] = p[j]; b[8 + j] = p[16 + j]; }
  return b;
}

// Async copy one 16B chunk global -> LDS (CDNA5 GLOBAL_LOAD_ASYNC_TO_LDS_B128,
// tracked by ASYNCcnt; no VGPR round-trip).
__device__ __forceinline__ void async_cp16(uint32_t lds_off, const void* gptr) {
  asm volatile("global_load_async_to_lds_b128 %0, %1, off"
               :: "v"(lds_off), "v"(gptr)
               : "memory");
}
__device__ __forceinline__ void wait_async0() {
  asm volatile("s_wait_asynccnt 0" ::: "memory");
}

// Low 32 bits of a generic pointer to LDS == LDS byte offset (ISA 10.2).
__device__ __forceinline__ uint32_t lds_off_of(const void* p) {
  return (uint32_t)(uintptr_t)p;
}

// ---------------------------------------------------------------------------
// Kernel 1 & 5: C[M=4096, N=1024] = A[M,1024] @ W[1024,1024] + bias
// Block tile 64(M) x 128(N), 8 waves, each wave 32x32 (2x2 WMMA tiles).
// IN_BF16: A is bf16 (concat-head activations) else f32.
// SPLIT_HEADS: write bf16 into [B,H,S,DH]; else write f32 [M,N].
// (These stages need f32->bf16 conversion, so they use the VGPR staging path.)
// ---------------------------------------------------------------------------
template <bool IN_BF16, bool SPLIT_HEADS>
__global__ __launch_bounds__(256) void gemm1024_kernel(
    const void* __restrict__ Ain, const float* __restrict__ W,
    const float* __restrict__ bias, void* __restrict__ Cout) {
  __shared__ bf16_t lds_a[64 * 40];    // 64 rows x 32 K, stride 40 (pad)
  __shared__ bf16_t lds_wt[128 * 40];  // W tile transposed: [n][k], stride 40

  const int tid = threadIdx.x;
  const int m0  = blockIdx.y * 64;
  const int n0  = blockIdx.x * 128;

  const int w    = tid >> 5;
  const int wm   = w & 1;          // wave grid 2(M) x 4(N)
  const int wn   = w >> 1;
  const int lane = tid & 31;
  const int ln   = lane & 15;
  const int mo   = (lane >> 4) * 8;

  v8f acc[2][2] = {};

  for (int k0 = 0; k0 < DM; k0 += 32) {
    // stage A 64x32 (convert to bf16)
#pragma unroll
    for (int i = 0; i < 8; ++i) {
      int idx = tid + i * 256;
      int r = idx >> 5, c = idx & 31;
      float fa;
      if (IN_BF16)
        fa = (float)((const bf16_t*)Ain)[(size_t)(m0 + r) * DM + k0 + c];
      else
        fa = ((const float*)Ain)[(size_t)(m0 + r) * DM + k0 + c];
      lds_a[r * 40 + c] = (bf16_t)fa;
    }
    // stage W 32x128, transposed into [n][k] so B-fragments are K-contiguous
#pragma unroll
    for (int i = 0; i < 16; ++i) {
      int idx = tid + i * 256;
      int kk = idx >> 7, nn = idx & 127;
      lds_wt[nn * 40 + kk] = (bf16_t)W[(size_t)(k0 + kk) * DM + n0 + nn];
    }
    if (k0 + 32 < DM)  // hint next W tile into cache (global_prefetch_b8)
      __builtin_prefetch(&W[(size_t)(k0 + 32) * DM + n0 + (tid & 127)], 0, 3);
    __syncthreads();

    v16bf a0 = ld_frag_a(lds_a + (wm * 32) * 40, 40);
    v16bf a1 = ld_frag_a(lds_a + (wm * 32 + 16) * 40, 40);
    v16bf b0 = ld_frag_bt(lds_wt + (wn * 32) * 40, 40);
    v16bf b1 = ld_frag_bt(lds_wt + (wn * 32 + 16) * 40, 40);
    acc[0][0] = wmma_bf16(a0, b0, acc[0][0]);
    acc[0][1] = wmma_bf16(a0, b1, acc[0][1]);
    acc[1][0] = wmma_bf16(a1, b0, acc[1][0]);
    acc[1][1] = wmma_bf16(a1, b1, acc[1][1]);
    __syncthreads();
  }

#pragma unroll
  for (int i = 0; i < 2; ++i) {
#pragma unroll
    for (int j = 0; j < 2; ++j) {
      const int col = n0 + wn * 32 + j * 16 + ln;
      const float bv = bias[col];
#pragma unroll
      for (int e = 0; e < 8; ++e) {
        const int row = m0 + wm * 32 + i * 16 + mo + e;
        const float val = acc[i][j][e] + bv;
        if (SPLIT_HEADS) {
          const int b = row >> 11, s = row & (SEQ - 1);
          const int h = col >> 6, d = col & (DH - 1);
          ((bf16_t*)Cout)[(((size_t)(b * NH + h)) * SEQ + s) * DH + d] =
              (bf16_t)val;
        } else {
          ((float*)Cout)[(size_t)row * DM + col] = val;
        }
      }
    }
  }
}

// ---------------------------------------------------------------------------
// Kernel 2: scores[b,h,q,k] = (Q . K) / sqrt(64), masked -> f32
// Block tile: 64 q x 128 k per (b,h); depth loop 64 = 2 WMMA K-steps.
// Q/K tiles are already bf16 in row-contiguous layout -> staged with CDNA5
// async global->LDS copies (GLOBAL_LOAD_ASYNC_TO_LDS_B128 + s_wait_asynccnt).
// ---------------------------------------------------------------------------
__global__ __launch_bounds__(256) void scores_kernel(
    const bf16_t* __restrict__ Qb, const bf16_t* __restrict__ Kb,
    const int* __restrict__ mask, float* __restrict__ attn) {
  __shared__ bf16_t lds_q[64 * 72];    // 64 q rows x 64 depth, stride 72
  __shared__ bf16_t lds_k[128 * 72];   // 128 k rows x 64 depth, stride 72

  const int tid = threadIdx.x;
  const int kt = blockIdx.x;   // 16 tiles of 128 keys
  const int qt = blockIdx.y;   // 32 tiles of 64 queries
  const int bh = blockIdx.z;   // B*H = 32
  const int b  = bh >> 4;

  const bf16_t* Qp = Qb + ((size_t)bh * SEQ + (size_t)qt * 64) * DH;
  const bf16_t* Kp = Kb + ((size_t)bh * SEQ + (size_t)kt * 128) * DH;

  // Row = 64 bf16 = 128 B = eight 16B chunks; LDS stride 144 B (16B aligned).
  const uint32_t q_base = lds_off_of(&lds_q[0]);
  const uint32_t k_base = lds_off_of(&lds_k[0]);
#pragma unroll
  for (int i = 0; i < 2; ++i) {        // Q: 512 chunks / 256 threads
    int c = tid + i * 256;
    int r = c >> 3, col = c & 7;
    async_cp16(q_base + (uint32_t)(r * 144 + col * 16), Qp + r * DH + col * 8);
  }
#pragma unroll
  for (int i = 0; i < 4; ++i) {        // K: 1024 chunks / 256 threads
    int c = tid + i * 256;
    int r = c >> 3, col = c & 7;
    async_cp16(k_base + (uint32_t)(r * 144 + col * 16), Kp + r * DH + col * 8);
  }
  wait_async0();
  __syncthreads();

  const int w = tid >> 5;
  const int wm = w & 1, wn = w >> 1;        // 2(q) x 4(k) waves
  const int lane = tid & 31;
  const int ln = lane & 15, mo = (lane >> 4) * 8;

  v8f acc[2][2] = {};
#pragma unroll
  for (int dk = 0; dk < DH; dk += 32) {
    v16bf a0 = ld_frag_a(lds_q + (wm * 32) * 72 + dk, 72);
    v16bf a1 = ld_frag_a(lds_q + (wm * 32 + 16) * 72 + dk, 72);
    v16bf b0 = ld_frag_bt(lds_k + (wn * 32) * 72 + dk, 72);
    v16bf b1 = ld_frag_bt(lds_k + (wn * 32 + 16) * 72 + dk, 72);
    acc[0][0] = wmma_bf16(a0, b0, acc[0][0]);
    acc[0][1] = wmma_bf16(a0, b1, acc[0][1]);
    acc[1][0] = wmma_bf16(a1, b0, acc[1][0]);
    acc[1][1] = wmma_bf16(a1, b1, acc[1][1]);
  }

  const float scale = 0.125f;  // 1/sqrt(DH)
  const size_t rowbase = (size_t)bh * SEQ * SEQ;
#pragma unroll
  for (int i = 0; i < 2; ++i) {
#pragma unroll
    for (int j = 0; j < 2; ++j) {
      const int kcol = kt * 128 + wn * 32 + j * 16 + ln;
#pragma unroll
      for (int e = 0; e < 8; ++e) {
        const int q = qt * 64 + wm * 32 + i * 16 + mo + e;
        const int mv = mask[((size_t)b * SEQ + q) * SEQ + kcol];
        const float s = acc[i][j][e] * scale;
        attn[rowbase + (size_t)q * SEQ + kcol] = (mv == 0) ? -1e9f : s;
      }
    }
  }
}

// ---------------------------------------------------------------------------
// Kernel 3: row softmax over 2048 elements, in place. One block per row.
// ---------------------------------------------------------------------------
__global__ __launch_bounds__(256) void softmax_kernel(float* __restrict__ attn) {
  __shared__ float red_max[8];
  __shared__ float red_sum[8];
  float* p = attn + (size_t)blockIdx.x * SEQ;
  const int tid = threadIdx.x;

  float vals[8];
  float mx = -INFINITY;
#pragma unroll
  for (int i = 0; i < 8; ++i) {
    vals[i] = p[tid + i * 256];
    mx = fmaxf(mx, vals[i]);
  }
#pragma unroll
  for (int off = 16; off > 0; off >>= 1)
    mx = fmaxf(mx, __shfl_xor(mx, off, 32));
  if ((tid & 31) == 0) red_max[tid >> 5] = mx;
  __syncthreads();
  float m2 = red_max[0];
#pragma unroll
  for (int i = 1; i < 8; ++i) m2 = fmaxf(m2, red_max[i]);

  float sum = 0.f;
#pragma unroll
  for (int i = 0; i < 8; ++i) {
    vals[i] = __expf(vals[i] - m2);
    sum += vals[i];
  }
#pragma unroll
  for (int off = 16; off > 0; off >>= 1) sum += __shfl_xor(sum, off, 32);
  if ((tid & 31) == 0) red_sum[tid >> 5] = sum;
  __syncthreads();
  float s2 = 0.f;
#pragma unroll
  for (int i = 0; i < 8; ++i) s2 += red_sum[i];
  const float inv = 1.0f / s2;
#pragma unroll
  for (int i = 0; i < 8; ++i) p[tid + i * 256] = vals[i] * inv;
}

// ---------------------------------------------------------------------------
// Kernel 4: out_h[b,h,q,d] = sum_k P[q,k] * V[k,d].
// Block tile 128 q x 64 d per (b,h); K loop over 2048 in steps of 32.
// P converted f32->bf16 while staging; V staged TRANSPOSED ([d][k]) so the
// B fragment reads are K-contiguous. Output written concat-head bf16.
// ---------------------------------------------------------------------------
__global__ __launch_bounds__(256) void pv_kernel(
    const float* __restrict__ attn, const bf16_t* __restrict__ Vb,
    bf16_t* __restrict__ Ocat) {
  __shared__ bf16_t lds_p[128 * 40];   // 128 q x 32 k, stride 40
  __shared__ bf16_t lds_vt[64 * 40];   // transposed V: [d][k], stride 40

  const int tid = threadIdx.x;
  const int qt = blockIdx.x;   // 16 tiles of 128 queries
  const int bh = blockIdx.y;   // 32
  const int b = bh >> 4, h = bh & 15;

  const float* Pp = attn + (size_t)bh * SEQ * SEQ + (size_t)qt * 128 * SEQ;
  const bf16_t* Vp = Vb + (size_t)bh * SEQ * DH;

  const int w = tid >> 5;
  const int wm = w & 3, wn = w >> 2;      // 4(q) x 2(d) waves
  const int lane = tid & 31;
  const int ln = lane & 15, mo = (lane >> 4) * 8;

  v8f acc[2][2] = {};

  for (int k0 = 0; k0 < SEQ; k0 += 32) {
#pragma unroll
    for (int i = 0; i < 16; ++i) {
      int idx = tid + i * 256;
      int r = idx >> 5, c = idx & 31;
      lds_p[r * 40 + c] = (bf16_t)Pp[(size_t)r * SEQ + k0 + c];
    }
#pragma unroll
    for (int i = 0; i < 8; ++i) {
      int idx = tid + i * 256;
      int kk = idx >> 6, d = idx & 63;
      lds_vt[d * 40 + kk] = Vp[(size_t)(k0 + kk) * DH + d];
    }
    if (k0 + 32 < SEQ)  // dominant 512MiB stream: prefetch next P tile
      __builtin_prefetch(&Pp[(size_t)(tid >> 1) * SEQ + k0 + 32 + (tid & 1) * 16],
                         0, 3);
    __syncthreads();

    v16bf a0 = ld_frag_a(lds_p + (wm * 32) * 40, 40);
    v16bf a1 = ld_frag_a(lds_p + (wm * 32 + 16) * 40, 40);
    v16bf b0 = ld_frag_bt(lds_vt + (wn * 32) * 40, 40);
    v16bf b1 = ld_frag_bt(lds_vt + (wn * 32 + 16) * 40, 40);
    acc[0][0] = wmma_bf16(a0, b0, acc[0][0]);
    acc[0][1] = wmma_bf16(a0, b1, acc[0][1]);
    acc[1][0] = wmma_bf16(a1, b0, acc[1][0]);
    acc[1][1] = wmma_bf16(a1, b1, acc[1][1]);
    __syncthreads();
  }

#pragma unroll
  for (int i = 0; i < 2; ++i) {
#pragma unroll
    for (int j = 0; j < 2; ++j) {
      const int d = wn * 32 + j * 16 + ln;
#pragma unroll
      for (int e = 0; e < 8; ++e) {
        const int q = qt * 128 + wm * 32 + i * 16 + mo + e;
        Ocat[((size_t)(b * SEQ + q)) * DM + h * DH + d] = (bf16_t)acc[i][j][e];
      }
    }
  }
}

// ---------------------------------------------------------------------------
// Launch
// ---------------------------------------------------------------------------
extern "C" void kernel_launch(void* const* d_in, const int* in_sizes, int n_in,
                              void* d_out, int out_size, void* d_ws,
                              size_t ws_size, hipStream_t stream) {
  const float* q    = (const float*)d_in[0];
  const float* k_   = (const float*)d_in[1];
  const float* v    = (const float*)d_in[2];
  const int*   mask = (const int*)d_in[3];
  const float* Wq = (const float*)d_in[4];
  const float* bq = (const float*)d_in[5];
  const float* Wk = (const float*)d_in[6];
  const float* bk = (const float*)d_in[7];
  const float* Wv = (const float*)d_in[8];
  const float* bv = (const float*)d_in[9];
  const float* Wo = (const float*)d_in[10];
  const float* bo = (const float*)d_in[11];

  // workspace: Qb | Kb | Vb | Ocat, each 4,194,304 bf16 (8 MiB) = 32 MiB
  const size_t per = (size_t)BQ * NH * SEQ * DH;
  bf16_t* Qb   = (bf16_t*)d_ws;
  bf16_t* Kb   = Qb + per;
  bf16_t* Vb   = Kb + per;
  bf16_t* Ocat = Vb + per;

  float* out  = (float*)d_out;                       // [B,S,D]
  float* attn = out + (size_t)BQ * SEQ * DM;         // [B,H,S,S]

  dim3 blk(256);
  dim3 g_gemm(DM / 128, (BQ * SEQ) / 64);            // (8, 64)

  // 1) projections -> bf16 split-head layout
  gemm1024_kernel<false, true><<<g_gemm, blk, 0, stream>>>(q, Wq, bq, Qb);
  gemm1024_kernel<false, true><<<g_gemm, blk, 0, stream>>>(k_, Wk, bk, Kb);
  gemm1024_kernel<false, true><<<g_gemm, blk, 0, stream>>>(v, Wv, bv, Vb);

  // 2) masked scores -> attention region (raw, pre-softmax)
  dim3 g_sc(SEQ / 128, SEQ / 64, BQ * NH);           // (16, 32, 32)
  scores_kernel<<<g_sc, blk, 0, stream>>>(Qb, Kb, mask, attn);

  // 3) row softmax in place
  softmax_kernel<<<dim3(BQ * NH * SEQ), blk, 0, stream>>>(attn);

  // 4) P @ V -> concat-head bf16
  dim3 g_pv(SEQ / 128, BQ * NH);                     // (16, 32)
  pv_kernel<<<g_pv, blk, 0, stream>>>(attn, Vb, Ocat);

  // 5) output projection -> f32 out
  gemm1024_kernel<true, false><<<g_gemm, blk, 0, stream>>>(Ocat, Wo, bo, out);
}
